// MultiHeadAttentionT_34127810134315
// MI455X (gfx1250) — matrix operand order repaired and checked by reading, not compiled, not run on hardware
//
#include <hip/hip_runtime.h>
#include <hip/hip_bf16.h>

// ---------------------------------------------------------------------------
// MHA forward for MI455X (gfx1250, wave32, WMMA 16x16x32 f16 -> f32 acc,
// TDM tensor_load_to_lds double-buffered staging of K and V^T).
// Outputs (concatenated in d_out): out[16,1024,19], attn[16,8,1024,1024],
// residual[16,1024,19], all f32.
// ---------------------------------------------------------------------------

#define B_SZ 16
#define L_SZ 1024
#define DM   19
#define NH   8
#define DH   64
#define HID  512
// 1/sqrt(512/9) = 3/sqrt(512)
#define INV_SCALE 0.13258252147247766f

typedef __attribute__((ext_vector_type(16))) _Float16 v16h;
typedef __attribute__((ext_vector_type(8)))  _Float16 v8h;
typedef __attribute__((ext_vector_type(8)))  float    v8f;
typedef __attribute__((ext_vector_type(4)))  unsigned int u32x4;
typedef __attribute__((ext_vector_type(8)))  int i32x8;
typedef __attribute__((ext_vector_type(4)))  int i32x4;

union V16 { v16h v; v8h h[2]; _Float16 e[16]; };

#if __has_builtin(__builtin_amdgcn_tensor_load_to_lds)
#define USE_TDM 1
#else
#define USE_TDM 0
#endif

#if USE_TDM
// 2-D TDM load: tile (tile_d1 rows x tile_d0 elems) of 2-byte elements from
// global (row stride stride0 elems) into LDS at byte offset lds_off, packed
// row-major. Descriptor bit layout per CDNA5 ISA ch. 8 (D# groups 0/1).
__device__ __forceinline__ void tdm_load_2d_f16(unsigned lds_off, const void* gaddr,
                                                unsigned tile_d0, unsigned tile_d1,
                                                unsigned stride0) {
  const unsigned long long ga = (unsigned long long)gaddr;
  u32x4 g0;
  g0[0] = 1u;                                             // count=1, user desc
  g0[1] = lds_off;                                        // lds_addr [63:32]
  g0[2] = (unsigned)(ga & 0xffffffffu);                   // global_addr lo
  g0[3] = (unsigned)((ga >> 32) & 0x01ffffffu) | (2u << 30); // addr hi | type=2
  i32x8 g1;
  g1[0] = (int)(1u << 16);                                // data_size=1 (2B)
  g1[1] = (int)((tile_d0 & 0xffffu) << 16);               // tensor_dim0 lo16
  g1[2] = (int)(((tile_d0 >> 16) & 0xffffu) |
                ((tile_d1 & 0xffffu) << 16));             // dim0 hi | dim1 lo
  g1[3] = (int)(((tile_d1 >> 16) & 0xffffu) |
                ((tile_d0 & 0xffffu) << 16));             // dim1 hi | tile_dim0
  g1[4] = (int)(tile_d1 & 0xffffu);                       // tile_dim1, tile_dim2=0
  g1[5] = (int)stride0;                                   // dim0_stride lo32
  g1[6] = 0;                                              // stride hi, dim1_stride lo
  g1[7] = 0;                                              // dim1_stride hi
  const i32x4 z4 = {0, 0, 0, 0};
#if defined(__clang_major__) && (__clang_major__ >= 23)
  const i32x8 z8 = {0, 0, 0, 0, 0, 0, 0, 0};
  __builtin_amdgcn_tensor_load_to_lds(g0, g1, z4, z4, z8, 0);
#else
  __builtin_amdgcn_tensor_load_to_lds(g0, g1, z4, z4, 0);
#endif
}
#endif

// ---------------------------------------------------------------------------
// Kernel 1: input projections (19 -> 512) x3, cast to f16, plus residual copy.
// q,k stored [B][H][L][64]; v stored transposed [B][H][64][L] so the P*V
// B-matrix (K x N = k-index x d) is lane-contiguous.
// ---------------------------------------------------------------------------
__global__ void proj_kernel(const float* __restrict__ Q, const float* __restrict__ K,
                            const float* __restrict__ V,
                            const float* __restrict__ Wq, const float* __restrict__ bq,
                            const float* __restrict__ Wk, const float* __restrict__ bk,
                            const float* __restrict__ Wv, const float* __restrict__ bv,
                            _Float16* __restrict__ qf, _Float16* __restrict__ kf,
                            _Float16* __restrict__ vt, float* __restrict__ res_out) {
  const int row = blockIdx.x;              // b*1024 + l
  const int b = row >> 10, l = row & 1023;
  __shared__ float xq[DM], xk[DM], xv[DM];
  const int t = threadIdx.x;               // 128 threads
  if (t < DM) {
    float q0 = Q[(size_t)row * DM + t];
    xq[t] = q0;
    xk[t] = K[(size_t)row * DM + t];
    xv[t] = V[(size_t)row * DM + t];
    res_out[(size_t)row * DM + t] = q0;    // residual output = Q
  }
  __syncthreads();
  #pragma unroll
  for (int c = 0; c < 4; ++c) {
    const int j = c * 128 + t;             // 0..511
    float aq = bq[j], ak = bk[j], av = bv[j];
    #pragma unroll
    for (int i = 0; i < DM; ++i) {
      aq += xq[i] * Wq[i * HID + j];
      ak += xk[i] * Wk[i * HID + j];
      av += xv[i] * Wv[i * HID + j];
    }
    const int h = j >> 6, d = j & 63;
    qf[(((size_t)(b * NH + h)) * L_SZ + l) * DH + d] = (_Float16)aq;
    kf[(((size_t)(b * NH + h)) * L_SZ + l) * DH + d] = (_Float16)ak;
    vt[(((size_t)(b * NH + h)) * DH + d) * L_SZ + l] = (_Float16)av;
  }
}

// ---------------------------------------------------------------------------
// Kernel 2: attention core. One WG = (b,h, 32 q-rows) vs all 1024 k.
// 256 threads = 8 waves. Dynamic LDS (~214 KB of 320 KB/WGP):
//   s_buf  f32 [32][1024]  (scores -> masked scores -> exp values)  131072 B
//   p_buf  f16 [32][1024]  (normalized probabilities)                65536 B
//   q_tile f16 [32][64]                                               4096 B
//   red    f32 [576]       (softmax partials)                         2304 B
//   kv_buf f16 2 x 4096    (TDM double buffer: K / V^T chunks)       16384 B
// ---------------------------------------------------------------------------
__global__ void attn_kernel(const _Float16* __restrict__ qf, const _Float16* __restrict__ kf,
                            const _Float16* __restrict__ vt,
                            const unsigned char* __restrict__ mask,
                            float* __restrict__ attn_out, float* __restrict__ ctx) {
  extern __shared__ char smem[];
  float*    s_buf  = (float*)smem;                         // [32][1024]
  _Float16* p_buf  = (_Float16*)(smem + 131072);           // [32][1024]
  _Float16* q_tile = (_Float16*)(smem + 196608);           // [32][64]
  float*    pmax   = (float*)(smem + 200704);              // [32*8]
  float*    psum   = pmax + 256;                           // [32*8]
  float*    rmax   = psum + 256;                           // [32]
  float*    rsum   = rmax + 32;                            // [32]
  _Float16* kv_buf = (_Float16*)(smem + 203008);           // 2 x [64][64] halves

  const int b  = blockIdx.x >> 3;
  const int h  = blockIdx.x & 7;
  const int q0 = blockIdx.y * 32;
  const int tid  = threadIdx.x;
  const int lane = tid & 31;
  const int w    = tid >> 5;       // wave 0..7
  const int m16  = lane & 15;
  const int grp  = lane >> 4;      // 0/1: lane half

  const _Float16* qbase = qf + ((size_t)(b * NH + h)) * L_SZ * DH;
  const _Float16* kbase = kf + ((size_t)(b * NH + h)) * L_SZ * DH;
  const _Float16* vbase = vt + ((size_t)(b * NH + h)) * DH * L_SZ;
  _Float16* kvb[2] = {kv_buf, kv_buf + 64 * DH};

  // ---- stage Q tile (32 rows x 64) into LDS ------------------------------
#if USE_TDM
  if (w == 0) {
    tdm_load_2d_f16((unsigned)(size_t)(void*)q_tile, qbase + (size_t)q0 * DH,
                    DH, 32, DH);
    // also pre-stage K chunk 0 (rows 0..63) for Phase 1
    tdm_load_2d_f16((unsigned)(size_t)(void*)kvb[0], kbase, DH, 64, DH);
    __builtin_amdgcn_s_wait_tensorcnt(0);
  }
  __syncthreads();
#else
  {
    const _Float16* src = qbase + (size_t)q0 * DH;
    for (int i = tid * 8; i < 32 * DH; i += 256 * 8) {
      *(v8h*)(q_tile + i) = *(const v8h*)(src + i);
    }
  }
  __syncthreads();
#endif

  // ---- Phase 1: S = (Q Kt) * invScale -> s_buf (mask deferred) ------------
  {
    const int qt = w & 1;          // which 16-row q tile
    const int g  = w >> 1;         // k-tile group 0..3 within a 64-row chunk
    // A matrix: q rows, 16x64 split into two K=32 chunks.
    // A k order used here (contiguous, matches B): half i of lane -> K = grp*16 + i
    V16 a0, a1;
    const _Float16* qrow = q_tile + (qt * 16 + m16) * DH;
    a0.h[0] = *(const v8h*)(qrow +  0 + grp * 16);
    a0.h[1] = *(const v8h*)(qrow +  8 + grp * 16);
    a1.h[0] = *(const v8h*)(qrow + 32 + grp * 16);
    a1.h[1] = *(const v8h*)(qrow + 40 + grp * 16);

    float* srow_base = s_buf + (qt * 16 + grp * 8) * 1024 + m16;

#if USE_TDM
    for (int t16 = 0; t16 < 16; ++t16) {
      // DMA next 64-row K chunk while computing on the current one
      if (w == 0 && t16 < 15) {
        tdm_load_2d_f16((unsigned)(size_t)(void*)kvb[(t16 + 1) & 1],
                        kbase + (size_t)(t16 + 1) * 64 * DH, DH, 64, DH);
      }
      const int kk0 = t16 * 64 + g * 16;        // k tile base column
      // B k order used here (contiguous, matches A): half i of lane -> K = (lane/16)*16 + i, N = lane%16
      const _Float16* kc = kvb[t16 & 1] + (g * 16 + m16) * DH;
      V16 b0, b1;
      b0.h[0] = *(const v8h*)(kc +  0 + grp * 16);
      b0.h[1] = *(const v8h*)(kc +  8 + grp * 16);
      b1.h[0] = *(const v8h*)(kc + 32 + grp * 16);
      b1.h[1] = *(const v8h*)(kc + 40 + grp * 16);

      v8f acc = {};
      acc = __builtin_amdgcn_wmma_f32_16x16x32_f16(false, a0.v, false, b0.v,
                                                   (short)0, acc, false, false);
      acc = __builtin_amdgcn_wmma_f32_16x16x32_f16(false, a1.v, false, b1.v,
                                                   (short)0, acc, false, false);
      // C layout: float i -> M = grp*8 + i, N = m16
      #pragma unroll
      for (int i = 0; i < 8; ++i) {
        srow_base[i * 1024 + kk0] = acc[i] * INV_SCALE;
      }
      if (w == 0) __builtin_amdgcn_s_wait_tensorcnt(0);
      __syncthreads();
    }
#else
    #pragma unroll 2
    for (int t16 = 0; t16 < 16; ++t16) {
      const int kk0 = t16 * 64 + g * 16;
      const _Float16* krow = kbase + (size_t)(kk0 + m16) * DH;
      if (t16 < 15) __builtin_prefetch(krow + 64 * DH, 0, 3);
      V16 b0, b1;
      b0.h[0] = *(const v8h*)(krow +  0 + grp * 16);
      b0.h[1] = *(const v8h*)(krow +  8 + grp * 16);
      b1.h[0] = *(const v8h*)(krow + 32 + grp * 16);
      b1.h[1] = *(const v8h*)(krow + 40 + grp * 16);
      v8f acc = {};
      acc = __builtin_amdgcn_wmma_f32_16x16x32_f16(false, a0.v, false, b0.v,
                                                   (short)0, acc, false, false);
      acc = __builtin_amdgcn_wmma_f32_16x16x32_f16(false, a1.v, false, b1.v,
                                                   (short)0, acc, false, false);
      #pragma unroll
      for (int i = 0; i < 8; ++i) {
        srow_base[i * 1024 + kk0] = acc[i] * INV_SCALE;
      }
    }
#endif
  }
  __syncthreads();

  // ---- Phase 2: mask + row softmax; attn (f32, global) + P (f16, LDS) -----
  {
    const int r  = tid & 31;          // row 0..31
    const int ch = tid >> 5;          // chunk 0..7 of 128 cols
    float* srow = s_buf + r * 1024 + ch * 128;
    const unsigned char* mrow =
        mask + ((size_t)b * L_SZ + (q0 + r)) * L_SZ + ch * 128;

    // pass 1: apply mask (vectorized b128 mask loads), write back, row max
    float mx = -3.4e38f;
    #pragma unroll
    for (int c = 0; c < 8; ++c) {
      const uint4 mw = *(const uint4*)(mrow + c * 16);
      const unsigned int wds[4] = {mw.x, mw.y, mw.z, mw.w};
      #pragma unroll
      for (int j = 0; j < 16; ++j) {
        float sv = srow[c * 16 + j];
        const bool mk = (wds[j >> 2] >> ((j & 3) * 8)) & 0xffu;
        sv = mk ? -1e9f : sv;
        srow[c * 16 + j] = sv;
        mx = fmaxf(mx, sv);
      }
    }
    pmax[r * 8 + ch] = mx;
    __syncthreads();
    if (tid < 32) {
      float m = pmax[tid * 8];
      #pragma unroll
      for (int c = 1; c < 8; ++c) m = fmaxf(m, pmax[tid * 8 + c]);
      rmax[tid] = m;
    }
    __syncthreads();

    // pass 2: e = exp(s - m), cache e in s_buf, partial sum
    const float m = rmax[r];
    float s = 0.f;
    for (int i = 0; i < 128; ++i) {
      const float e = __expf(srow[i] - m);
      srow[i] = e;
      s += e;
    }
    psum[r * 8 + ch] = s;
    __syncthreads();
    if (tid < 32) {
      float t = 0.f;
      #pragma unroll
      for (int c = 0; c < 8; ++c) t += psum[tid * 8 + c];
      rsum[tid] = t;
    }
    __syncthreads();

    // pass 3: p = e * inv -> global attn (float4) + LDS f16
    const float inv = 1.f / rsum[r];
    _Float16* prow = p_buf + r * 1024 + ch * 128;
    float* arow = attn_out +
        (((size_t)(b * NH + h)) * L_SZ + (q0 + r)) * L_SZ + ch * 128;
    for (int i = 0; i < 128; i += 4) {
      const float4 ev = *(const float4*)(srow + i);
      float4 pv;
      pv.x = ev.x * inv;
      pv.y = ev.y * inv;
      pv.z = ev.z * inv;
      pv.w = ev.w * inv;
      prow[i + 0] = (_Float16)pv.x;
      prow[i + 1] = (_Float16)pv.y;
      prow[i + 2] = (_Float16)pv.z;
      prow[i + 3] = (_Float16)pv.w;
      *(float4*)(arow + i) = pv;
    }
  }
  __syncthreads();

  // ---- Phase 3: context = P V; TDM-staged V^T chunks, 2 acc chains --------
  {
    const int qt     = w & 1;
    const int dchunk = w >> 1;        // 0..3 -> 16 d columns each
    const int dcol   = dchunk * 16 + m16;
    const _Float16* prow_base = p_buf + (qt * 16 + m16) * 1024;
    v8f acc0 = {};
    v8f acc1 = {};
#if USE_TDM
    // chunk c: V^T rows d=0..63, k columns [c*64, c*64+64), packed [64][64]
    if (w == 0) {
      tdm_load_2d_f16((unsigned)(size_t)(void*)kvb[0], vbase, 64, DH, L_SZ);
      __builtin_amdgcn_s_wait_tensorcnt(0);
    }
    __syncthreads();
    for (int c = 0; c < 16; ++c) {
      if (w == 0 && c < 15) {
        tdm_load_2d_f16((unsigned)(size_t)(void*)kvb[(c + 1) & 1],
                        vbase + (c + 1) * 64, 64, DH, L_SZ);
      }
      const _Float16* vchunk = kvb[c & 1] + dcol * 64;
      #pragma unroll
      for (int s = 0; s < 2; ++s) {
        const int col = c * 64 + s * 32;
        V16 a, bm;
        const _Float16* pr = prow_base + col;
        a.h[0] = *(const v8h*)(pr +  0 + grp * 16);
        a.h[1] = *(const v8h*)(pr +  8 + grp * 16);
        const _Float16* vc = vchunk + s * 32 + grp * 16;
        bm.h[0] = *(const v8h*)(vc + 0);
        bm.h[1] = *(const v8h*)(vc + 8);
        if (s == 0)
          acc0 = __builtin_amdgcn_wmma_f32_16x16x32_f16(false, a.v, false, bm.v,
                                                        (short)0, acc0, false, false);
        else
          acc1 = __builtin_amdgcn_wmma_f32_16x16x32_f16(false, a.v, false, bm.v,
                                                        (short)0, acc1, false, false);
      }
      if (w == 0) __builtin_amdgcn_s_wait_tensorcnt(0);
      __syncthreads();
    }
#else
    const _Float16* vrowb = vbase + (size_t)dcol * L_SZ;
    #pragma unroll 2
    for (int jj = 0; jj < 32; jj += 2) {
      V16 a0, b0, a1, b1;
      const _Float16* pr0 = prow_base + jj * 32;
      a0.h[0] = *(const v8h*)(pr0 +  0 + grp * 16);
      a0.h[1] = *(const v8h*)(pr0 +  8 + grp * 16);
      a1.h[0] = *(const v8h*)(pr0 + 32 + grp * 16);
      a1.h[1] = *(const v8h*)(pr0 + 40 + grp * 16);
      const _Float16* vr0 = vrowb + jj * 32 + grp * 16;
      if (jj < 30) __builtin_prefetch(vr0 + 64, 0, 3);
      b0.h[0] = *(const v8h*)(vr0 + 0);
      b0.h[1] = *(const v8h*)(vr0 + 8);
      b1.h[0] = *(const v8h*)(vr0 + 32);
      b1.h[1] = *(const v8h*)(vr0 + 40);
      acc0 = __builtin_amdgcn_wmma_f32_16x16x32_f16(false, a0.v, false, b0.v,
                                                    (short)0, acc0, false, false);
      acc1 = __builtin_amdgcn_wmma_f32_16x16x32_f16(false, a1.v, false, b1.v,
                                                    (short)0, acc1, false, false);
    }
#endif
    const v8f acc = acc0 + acc1;
    #pragma unroll
    for (int i = 0; i < 8; ++i) {
      const int row = q0 + qt * 16 + grp * 8 + i;
      ctx[((size_t)b * L_SZ + row) * HID + h * DH + dcol] = acc[i];
    }
  }
}

// ---------------------------------------------------------------------------
// Kernel 3: out = LayerNorm(ctx @ Wo + bo + residual). One 64-thread block/row.
// ---------------------------------------------------------------------------
__global__ void out_ln_kernel(const float* __restrict__ ctx, const float* __restrict__ Wo,
                              const float* __restrict__ bo, const float* __restrict__ Qin,
                              const float* __restrict__ g, const float* __restrict__ beta,
                              float* __restrict__ out) {
  const int row = blockIdx.x;        // b*1024 + l
  __shared__ float cx[HID];
  __shared__ float y[DM];
  __shared__ float mu_s, rv_s;
  const int t = threadIdx.x;         // 64 threads
  for (int i = t; i < HID; i += 64) cx[i] = ctx[(size_t)row * HID + i];
  __syncthreads();
  if (t < DM) {
    float acc = bo[t];
    for (int i = 0; i < HID; ++i) acc += cx[i] * Wo[i * DM + t];
    y[t] = acc + Qin[(size_t)row * DM + t];
  }
  __syncthreads();
  if (t == 0) {
    float mu = 0.f;
    #pragma unroll
    for (int i = 0; i < DM; ++i) mu += y[i];
    mu *= (1.f / DM);
    float var = 0.f;
    #pragma unroll
    for (int i = 0; i < DM; ++i) { float d = y[i] - mu; var += d * d; }
    var *= (1.f / DM);
    mu_s = mu;
    rv_s = rsqrtf(var + 1e-5f);
  }
  __syncthreads();
  if (t < DM) out[(size_t)row * DM + t] = (y[t] - mu_s) * rv_s * g[t] + beta[t];
}

// ---------------------------------------------------------------------------
extern "C" void kernel_launch(void* const* d_in, const int* in_sizes, int n_in,
                              void* d_out, int out_size, void* d_ws, size_t ws_size,
                              hipStream_t stream) {
  const float* Q  = (const float*)d_in[0];
  const float* K  = (const float*)d_in[1];
  const float* V  = (const float*)d_in[2];
  const unsigned char* mask = (const unsigned char*)d_in[3];   // jax bool = 1 byte
  const float* Wq = (const float*)d_in[4];
  const float* bq = (const float*)d_in[5];
  const float* Wk = (const float*)d_in[6];
  const float* bk = (const float*)d_in[7];
  const float* Wv = (const float*)d_in[8];
  const float* bv = (const float*)d_in[9];
  const float* Wo = (const float*)d_in[10];
  const float* bo = (const float*)d_in[11];
  const float* ln_g = (const float*)d_in[12];
  const float* ln_b = (const float*)d_in[13];

  float* out_p  = (float*)d_out;                               // [16,1024,19]
  float* attn_p = out_p + (size_t)B_SZ * L_SZ * DM;            // [16,8,1024,1024]
  float* res_p  = attn_p + (size_t)B_SZ * NH * L_SZ * L_SZ;    // [16,1024,19]

  // Workspace layout (needs ~84 MB): qf16, kf16, vt_f16, ctx_f32.
  char* ws = (char*)d_ws;
  const size_t half_mat = (size_t)B_SZ * NH * L_SZ * DH * sizeof(_Float16); // 16 MB
  _Float16* qf = (_Float16*)ws;
  _Float16* kf = (_Float16*)(ws + half_mat);
  _Float16* vt = (_Float16*)(ws + 2 * half_mat);
  float*    ctx = (float*)(ws + 3 * half_mat);

  const int smem_bytes = 219392;  // ~214 KB of the 320 KB/WGP LDS
  hipFuncSetAttribute((const void*)attn_kernel,
                      hipFuncAttributeMaxDynamicSharedMemorySize, smem_bytes);

  proj_kernel<<<dim3(B_SZ * L_SZ), 128, 0, stream>>>(Q, K, V, Wq, bq, Wk, bk,
                                                     Wv, bv, qf, kf, vt, res_p);
  attn_kernel<<<dim3(B_SZ * NH, L_SZ / 32), 256, smem_bytes, stream>>>(
      qf, kf, vt, mask, attn_p, ctx);
  out_ln_kernel<<<dim3(B_SZ * L_SZ), 64, 0, stream>>>(ctx, Wo, bo, Q, ln_g,
                                                      ln_b, out_p);
}
